// LanguageModelFinalWrapper_42468636623331
// MI455X (gfx1250) — compile-verified, matrix operand order checked by reading, not run on hardware
//
#include <hip/hip_runtime.h>
#include <hip/hip_bf16.h>

typedef __bf16 bf16_t;
typedef __attribute__((ext_vector_type(16))) __bf16 v16bf;
typedef __attribute__((ext_vector_type(4)))  __bf16 v4bf;
typedef __attribute__((ext_vector_type(8)))  float  v8f;
typedef __attribute__((ext_vector_type(4)))  float  v4f;

constexpr int kB = 2;
constexpr int kS = 2048;
constexpr int kD = 4096;
constexpr int kV = 32000;
constexpr int kM = kB * kS;          // 4096 rows of logits
constexpr float kEps = 1e-6f;

// ---------------------------------------------------------------------------
// Kernel 1: RMSNorm fp32 [M, D] -> bf16 [M, D] (workspace).
// One block (256 threads) per row; each thread handles 16 elements.
// ---------------------------------------------------------------------------
__global__ __launch_bounds__(256) void rmsnorm_to_bf16_kernel(
    const float* __restrict__ x, const float* __restrict__ w,
    bf16_t* __restrict__ out)
{
    const int row = blockIdx.x;
    const float* xr = x + (size_t)row * kD;
    bf16_t* orow = out + (size_t)row * kD;

    float acc = 0.f;
#pragma unroll
    for (int i = 0; i < 4; ++i) {
        v4f v = *(const v4f*)(xr + threadIdx.x * 4 + i * 1024);
        acc += v.x * v.x + v.y * v.y + v.z * v.z + v.w * v.w;
    }

    __shared__ float red[256];
    red[threadIdx.x] = acc;
    __syncthreads();
#pragma unroll
    for (int off = 128; off > 0; off >>= 1) {
        if (threadIdx.x < off) red[threadIdx.x] += red[threadIdx.x + off];
        __syncthreads();
    }
    const float inv = rsqrtf(red[0] * (1.0f / kD) + kEps);

#pragma unroll
    for (int i = 0; i < 4; ++i) {
        const int idx = threadIdx.x * 4 + i * 1024;
        v4f v  = *(const v4f*)(xr + idx);
        v4f wv = *(const v4f*)(w + idx);
        v4bf o;
        o[0] = (bf16_t)(v.x * inv * wv.x);
        o[1] = (bf16_t)(v.y * inv * wv.y);
        o[2] = (bf16_t)(v.z * inv * wv.z);
        o[3] = (bf16_t)(v.w * inv * wv.w);
        *(v4bf*)(orow + idx) = o;
    }
}

// ---------------------------------------------------------------------------
// Kernel 2: logits = Xn[M, D](bf16) * W[V, D](fp32, cvt->bf16 on the fly)^T
// Block = 8 waves = 64M x 128N tile. Wave tile = 16M x 64N (4 accumulators).
// Epilogue applies the label-shift row mapping and uses non-temporal stores.
// ---------------------------------------------------------------------------
__device__ __forceinline__ v16bf load_cvt_bf16(const float* __restrict__ p)
{
    v16bf r;
#pragma unroll
    for (int i = 0; i < 4; ++i) {
        v4f v = *(const v4f*)(p + 4 * i);
        r[4 * i + 0] = (bf16_t)v.x;
        r[4 * i + 1] = (bf16_t)v.y;
        r[4 * i + 2] = (bf16_t)v.z;
        r[4 * i + 3] = (bf16_t)v.w;
    }
    return r;
}

__global__ __launch_bounds__(256) void lmhead_wmma_kernel(
    const bf16_t* __restrict__ Xn, const float* __restrict__ W,
    float* __restrict__ out)
{
    const int lane = threadIdx.x & 31;
    const int wave = threadIdx.x >> 5;
    const int wm = wave & 3;                 // 4 waves along M
    const int wn = wave >> 2;                // 2 waves along N

    // blockIdx.x walks M (fast) so consecutive blocks share the same weight
    // tile -> each weight byte is fetched from HBM once, reused via L2.
    const int m0 = blockIdx.x * 64 + wm * 16;
    const int n0 = blockIdx.y * 128 + wn * 64;

    const int lrow  = lane & 15;             // row within 16-row fragment
    const int khalf = (lane >> 4) * 16;      // contiguous K half: k = 16*h + s, same in A and B

    const bf16_t* Ap  = Xn + (size_t)(m0 + lrow) * kD + khalf;
    const float*  Bp0 = W + (size_t)(n0 +  0 + lrow) * kD + khalf;
    const float*  Bp1 = W + (size_t)(n0 + 16 + lrow) * kD + khalf;
    const float*  Bp2 = W + (size_t)(n0 + 32 + lrow) * kD + khalf;
    const float*  Bp3 = W + (size_t)(n0 + 48 + lrow) * kD + khalf;

    v8f acc0 = {}, acc1 = {}, acc2 = {}, acc3 = {};

    for (int k = 0; k < kD; k += 32) {
        const v16bf a  = *(const v16bf*)(Ap + k);
        const v16bf b0 = load_cvt_bf16(Bp0 + k);
        const v16bf b1 = load_cvt_bf16(Bp1 + k);
        const v16bf b2 = load_cvt_bf16(Bp2 + k);
        const v16bf b3 = load_cvt_bf16(Bp3 + k);
        acc0 = __builtin_amdgcn_wmma_f32_16x16x32_bf16(false, a, false, b0,
                                                       (short)0, acc0, false, false);
        acc1 = __builtin_amdgcn_wmma_f32_16x16x32_bf16(false, a, false, b1,
                                                       (short)0, acc1, false, false);
        acc2 = __builtin_amdgcn_wmma_f32_16x16x32_bf16(false, a, false, b2,
                                                       (short)0, acc2, false, false);
        acc3 = __builtin_amdgcn_wmma_f32_16x16x32_bf16(false, a, false, b3,
                                                       (short)0, acc3, false, false);
    }

    // C/D fragment layout: VGPR i, lanes 0-15 -> (M = i,   N = lane),
    //                               lanes 16-31 -> (M = i+8, N = lane-16).
    const int ncol  = lane & 15;
    const int mrow0 = m0 + 8 * (lane >> 4);

    const v8f accs[4] = {acc0, acc1, acc2, acc3};
#pragma unroll
    for (int nb = 0; nb < 4; ++nb) {
        const int n = n0 + nb * 16 + ncol;
#pragma unroll
        for (int i = 0; i < 8; ++i) {
            const int m = mrow0 + i;
            const int s = m & (kS - 1);          // m % 2048
            if (s != kS - 1) {                    // shift: drop last position
                const int b = m >> 11;            // m / 2048
                const size_t orow = (size_t)(b * (kS - 1) + s);
                __builtin_nontemporal_store(accs[nb][i], out + orow * kV + n);
            }
        }
    }
}

// ---------------------------------------------------------------------------
// Kernel 3: shifted labels (int64) appended after logits, as fp32 values.
// ---------------------------------------------------------------------------
__global__ __launch_bounds__(256) void shift_labels_kernel(
    const long long* __restrict__ labels, float* __restrict__ out)
{
    const int n = kB * (kS - 1);                 // 4094
    const int i = blockIdx.x * 256 + threadIdx.x;
    if (i < n) {
        const int b = i / (kS - 1);
        const int s = i - b * (kS - 1);
        out[i] = (float)labels[(size_t)b * kS + s + 1];
    }
}

// ---------------------------------------------------------------------------
extern "C" void kernel_launch(void* const* d_in, const int* in_sizes, int n_in,
                              void* d_out, int out_size, void* d_ws, size_t ws_size,
                              hipStream_t stream)
{
    const float*     hidden = (const float*)d_in[0];
    const float*     nw     = (const float*)d_in[1];
    const float*     lmw    = (const float*)d_in[2];
    // d_in[3] = attention_mask (all ones; no effect on the static-shape path)
    const long long* labels = (const long long*)d_in[4];
    float* out = (float*)d_out;

    bf16_t* xn = (bf16_t*)d_ws;                  // 4096*4096*2 = 32 MB scratch

    rmsnorm_to_bf16_kernel<<<kM, 256, 0, stream>>>(hidden, nw, xn);

    dim3 grid(kM / 64, kV / 128);                // (64, 250); x walks M
    lmhead_wmma_kernel<<<grid, 256, 0, stream>>>(xn, lmw, out);

    const size_t logit_count = (size_t)kB * (kS - 1) * kV;
    shift_labels_kernel<<<(kB * (kS - 1) + 255) / 256, 256, 0, stream>>>(
        labels, out + logit_count);
}
